// POS_predictor_20804821582149
// MI455X (gfx1250) — matrix operand order repaired and checked by reading, not compiled, not run on hardware
//
#include <hip/hip_runtime.h>

typedef __attribute__((ext_vector_type(16))) __bf16 v16bf;
typedef __attribute__((ext_vector_type(8)))  float  v8f;

#define TT   256   // timesteps
#define BB   64    // batch
#define EE   256   // embedding dim
#define HH   512   // hidden
#define GG   1536  // 3*H
#define KCAT 768   // H + E
#define NC   20    // classes

__device__ __forceinline__ unsigned short f2bf(float f) {
  unsigned int u = __float_as_uint(f);
  return (unsigned short)((u + 0x7FFFu + ((u >> 16) & 1u)) >> 16);
}

__device__ __forceinline__ float sigm(float x) {
  return 1.0f / (1.0f + __expf(-x));
}

__device__ __forceinline__ float tanh_fast(float x) {
  x = fminf(fmaxf(x, -15.0f), 15.0f);
  float e = __expf(2.0f * x);
  return (e - 1.0f) / (e + 1.0f);
}

union Frag {
  v16bf v;
  uint4 q[2];
};

// ---------------- prep kernels ----------------

// xe_bf16[t*B+b][e] = bf16(emb[x[b][t]][e])
__global__ void k_gather_xe(const int* __restrict__ x, const float* __restrict__ emb,
                            unsigned short* __restrict__ xe) {
  size_t i = (size_t)blockIdx.x * blockDim.x + threadIdx.x;
  if (i >= (size_t)TT * BB * EE) return;
  int e   = (int)(i & (EE - 1));
  int row = (int)(i >> 8);           // t*B + b
  int t   = row >> 6;
  int b   = row & 63;
  int tok = x[b * TT + t];
  xe[i] = f2bf(emb[(size_t)tok * EE + e]);
}

// W_cat[g][k] : k<512 -> W_hh[g][k] ; else W_ih[g][k-512]
__global__ void k_wcat(const float* __restrict__ Wih, const float* __restrict__ Whh,
                       unsigned short* __restrict__ wcat) {
  size_t i = (size_t)blockIdx.x * blockDim.x + threadIdx.x;
  if (i >= (size_t)GG * KCAT) return;
  int g = (int)(i / KCAT);
  int k = (int)(i % KCAT);
  float v = (k < HH) ? Whh[(size_t)g * HH + k] : Wih[(size_t)g * EE + (k - HH)];
  wcat[i] = f2bf(v);
}

// W_cls padded to [32][512] bf16
__global__ void k_wcls(const float* __restrict__ Wcls, unsigned short* __restrict__ wcls) {
  size_t i = (size_t)blockIdx.x * blockDim.x + threadIdx.x;
  if (i >= (size_t)32 * HH) return;
  int r = (int)(i >> 9);
  int k = (int)(i & 511);
  wcls[i] = (r < NC) ? f2bf(Wcls[(size_t)r * HH + k]) : (unsigned short)0;
}

// zero h state slice 0 (f32 + bf16)
__global__ void k_hzero(float* __restrict__ hsf, unsigned short* __restrict__ hsbf) {
  size_t i = (size_t)blockIdx.x * blockDim.x + threadIdx.x;
  if (i >= (size_t)BB * HH) return;
  hsf[i]  = 0.0f;
  hsbf[i] = 0;
}

// ---------------- GRU step: one launch per timestep t = 1..256 ----------------
// 128 waves total as 16 blocks x 8 waves. Within a block, wave w:
//   bi = w & 3  (batch tile)  -- the 4 bi-waves share IDENTICAL B fragments,
//   hj = blockIdx*2 + (w>>2)  -- so W_cat tiles are fetched once per WGP (L0 reuse).
__global__ __launch_bounds__(256) void k_gru_step(
    int t,
    const unsigned short* __restrict__ xe,    // [T][B][E] bf16
    const unsigned short* __restrict__ wcat,  // [1536][768] bf16
    const float* __restrict__ b_ih, const float* __restrict__ b_hh,
    unsigned short* __restrict__ hsbf,        // [T+1][B][H] bf16
    float* __restrict__ hsf)                  // [T+1][B][H] f32
{
  const int lane = threadIdx.x & 31;
  const int wv   = threadIdx.x >> 5;           // 0..7
  const int bi   = wv & 3;                     // 0..3
  const int hj   = (blockIdx.x << 1) + (wv >> 2); // 0..31
  const int b0   = bi << 4;
  const int n0   = hj << 4;
  const int half = lane >> 4;
  const int lrow = lane & 15;

  v8f acc_r = {}; v8f acc_z = {}; v8f acc_hn = {}; v8f acc_in = {};

  const unsigned short* harow = hsbf + (size_t)(t - 1) * (BB * HH) + (size_t)(b0 + lrow) * HH;
  const unsigned short* xrow  = xe   + ((size_t)(t - 1) * BB + (b0 + lrow)) * EE;

  const unsigned short* wr = wcat + (size_t)(n0 +          lrow) * KCAT;
  const unsigned short* wz = wcat + (size_t)(n0 + HH     + lrow) * KCAT;
  const unsigned short* wn = wcat + (size_t)(n0 + 2 * HH + lrow) * KCAT;

#pragma unroll
  for (int kt = 0; kt < 24; ++kt) {
    const int k0 = kt << 5;
    const int kb = k0 + (half << 4);     // A: 16 contiguous K values per lane
    Frag a;
    if (k0 < HH) {
      a.q[0] = *(const uint4*)(harow + kb);
      a.q[1] = *(const uint4*)(harow + kb + 8);
    } else {
      a.q[0] = *(const uint4*)(xrow + (kb - HH));
      a.q[1] = *(const uint4*)(xrow + (kb - HH) + 8);
    }
    const int kbB = k0 + (half << 4);    // B: 16 contiguous K values per lane
    Frag br, bz, bn;
    br.q[0] = *(const uint4*)(wr + kbB); br.q[1] = *(const uint4*)(wr + kbB + 8);
    bz.q[0] = *(const uint4*)(wz + kbB); bz.q[1] = *(const uint4*)(wz + kbB + 8);
    bn.q[0] = *(const uint4*)(wn + kbB); bn.q[1] = *(const uint4*)(wn + kbB + 8);

    acc_r = __builtin_amdgcn_wmma_f32_16x16x32_bf16(false, a.v, false, br.v, (short)0, acc_r, false, false);
    acc_z = __builtin_amdgcn_wmma_f32_16x16x32_bf16(false, a.v, false, bz.v, (short)0, acc_z, false, false);
    if (k0 < HH)
      acc_hn = __builtin_amdgcn_wmma_f32_16x16x32_bf16(false, a.v, false, bn.v, (short)0, acc_hn, false, false);
    else
      acc_in = __builtin_amdgcn_wmma_f32_16x16x32_bf16(false, a.v, false, bn.v, (short)0, acc_in, false, false);
  }

  const int col = n0 + lrow;                // D col N = lane%16
  const float brz = b_ih[col]          + b_hh[col];
  const float bzz = b_ih[col + HH]     + b_hh[col + HH];
  const float bin = b_ih[col + 2 * HH];
  const float bhn = b_hh[col + 2 * HH];

  const float*    hprev_f = hsf  + (size_t)(t - 1) * (BB * HH);
  float*          hnew_f  = hsf  + (size_t)t * (BB * HH);
  unsigned short* hnew_b  = hsbf + (size_t)t * (BB * HH);

#pragma unroll
  for (int v = 0; v < 8; ++v) {
    const int brow = b0 + v + (half << 3);  // D row M = v + 8*(lane/16)
    const float r  = sigm(acc_r[v] + brz);
    const float z  = sigm(acc_z[v] + bzz);
    const float nn = tanh_fast((acc_in[v] + bin) + r * (acc_hn[v] + bhn));
    const float hp = hprev_f[(size_t)brow * HH + col];
    const float hv = (1.0f - z) * nn + z * hp;
    hnew_f[(size_t)brow * HH + col] = hv;
    hnew_b[(size_t)brow * HH + col] = f2bf(hv);
  }
}

// ---------------- classifier: out[b][t][c] = hs[t][b] . W_cls[c] + b_cls[c] ----------------
__global__ __launch_bounds__(256) void k_cls(
    const unsigned short* __restrict__ hsbf,   // [T+1][B][H] bf16 (slices 1..T)
    const unsigned short* __restrict__ wcls,   // [32][512] bf16
    const float* __restrict__ b_cls,
    float* __restrict__ out)                   // [B][T][20] f32
{
  const int lane = threadIdx.x & 31;
  const int wid  = blockIdx.x * (blockDim.x >> 5) + (threadIdx.x >> 5); // 0..2047
  const int ct   = wid & 1;           // col tile (classes 0-15 / 16-31)
  const int rest = wid >> 1;
  const int b    = rest >> 4;         // 0..63
  const int t0   = (rest & 15) << 4;  // 0,16,...,240
  const int half = lane >> 4;
  const int lrow = lane & 15;
  const int c0   = ct << 4;

  v8f acc = {};
  const unsigned short* arow = hsbf + ((size_t)(t0 + lrow + 1) * BB + b) * HH;
  const unsigned short* brow = wcls + (size_t)(c0 + lrow) * HH;

#pragma unroll
  for (int kt = 0; kt < 16; ++kt) {
    const int k0 = kt << 5;
    const int kb = k0 + (half << 4);
    Frag a, bf;
    a.q[0]  = *(const uint4*)(arow + kb);
    a.q[1]  = *(const uint4*)(arow + kb + 8);
    const int kbB = k0 + (half << 4);
    bf.q[0] = *(const uint4*)(brow + kbB);
    bf.q[1] = *(const uint4*)(brow + kbB + 8);
    acc = __builtin_amdgcn_wmma_f32_16x16x32_bf16(false, a.v, false, bf.v, (short)0, acc, false, false);
  }

  const int col = c0 + lrow;
  if (col < NC) {
    const float bc = b_cls[col];
#pragma unroll
    for (int v = 0; v < 8; ++v) {
      const int tt = t0 + v + (half << 3);
      out[((size_t)b * TT + tt) * NC + col] = acc[v] + bc;
    }
  }
}

// ---------------- launcher ----------------
extern "C" void kernel_launch(void* const* d_in, const int* in_sizes, int n_in,
                              void* d_out, int out_size, void* d_ws, size_t ws_size,
                              hipStream_t stream) {
  const int*   x     = (const int*)  d_in[0];
  const float* emb   = (const float*)d_in[1];
  const float* W_ih  = (const float*)d_in[2];
  const float* W_hh  = (const float*)d_in[3];
  const float* b_ih  = (const float*)d_in[4];
  const float* b_hh  = (const float*)d_in[5];
  const float* W_cls = (const float*)d_in[6];
  const float* b_cls = (const float*)d_in[7];
  float* out = (float*)d_out;

  char* w = (char*)d_ws;
  // workspace carve (all 256B aligned)
  unsigned short* xe   = (unsigned short*)(w);                       //  8,388,608 B
  unsigned short* wcat = (unsigned short*)(w + 8388608);             //  2,359,296 B
  unsigned short* wcls = (unsigned short*)(w + 8388608 + 2359296);   //     32,768 B
  unsigned short* hsbf = (unsigned short*)(w + 10780672);            // 16,842,752 B
  float*          hsf  = (float*)         (w + 27623424);            // 33,685,504 B
  (void)ws_size; (void)in_sizes; (void)n_in; (void)out_size;

  // prep
  {
    size_t n = (size_t)TT * BB * EE;                 // 4,194,304
    k_gather_xe<<<(unsigned)((n + 255) / 256), 256, 0, stream>>>(x, emb, xe);
  }
  {
    size_t n = (size_t)GG * KCAT;                    // 1,179,648
    k_wcat<<<(unsigned)((n + 255) / 256), 256, 0, stream>>>(W_ih, W_hh, wcat);
  }
  k_wcls<<<64, 256, 0, stream>>>(W_cls, wcls);       // 16,384
  k_hzero<<<128, 256, 0, stream>>>(hsf, hsbf);       // 32,768

  // serial GRU scan: 256 stream-ordered step launches (graph-capture safe)
  for (int t = 1; t <= TT; ++t) {
    k_gru_step<<<16, 256, 0, stream>>>(t, xe, wcat, b_ih, b_hh, hsbf, hsf);
  }

  // classifier GEMM over all (b,t): 2048 waves
  k_cls<<<256, 256, 0, stream>>>(hsbf, wcls, b_cls, out);
}
